// GraphSAGE_8761733284693
// MI455X (gfx1250) — matrix operand order repaired and checked by reading, not compiled, not run on hardware
//
#include <hip/hip_runtime.h>

// GraphSAGE (2x SAGEConv mean-aggr) for MI455X / gfx1250.
// Phase 1: edge scatter (L2-resident float atomics).
// Phase 2: fused mean + dual-GEMM (bf16 WMMA, fp32 accum), weight-stationary:
//          each wave pins its B (weight) fragments in VGPRs and streams 8 row
//          tiles through them, prefetching the next tile's A rows.

#define N_NODES   50000
#define N_EDGES   800000
#define DIM_IN    128
#define DIM_HID   128
#define DIM_OUT   40
#define ROW_TILES 3125              // 50000 / 16
#define ROWS_PER_WAVE 8

typedef __attribute__((ext_vector_type(16))) __bf16 v16bf;
typedef __attribute__((ext_vector_type(8)))  float  v8f;

// ---------------- Phase 1: degree counts + feature scatter-add ----------------

__global__ void count_kernel(const long long* __restrict__ dst,
                             float* __restrict__ counts) {
    int e = blockIdx.x * blockDim.x + threadIdx.x;
    if (e < N_EDGES) atomicAdd(&counts[(int)dst[e]], 1.0f);
}

// One thread per (edge, 4-float chunk): E*32 threads. x and the accumulator
// both fit in the 192MB L2, so these atomics stay on-chip.
__global__ void scatter_kernel(const long long* __restrict__ src,
                               const long long* __restrict__ dst,
                               const float* __restrict__ feat,
                               float* __restrict__ acc) {
    int t = blockIdx.x * blockDim.x + threadIdx.x;
    if (t >= N_EDGES * 32) return;
    int e = t >> 5;
    int q = (t & 31) << 2;
    int s = (int)src[e];
    int d = (int)dst[e];
    const float4 v = *reinterpret_cast<const float4*>(feat + s * DIM_IN + q);
    float* p = acc + d * DIM_IN + q;
    atomicAdd(p + 0, v.x);
    atomicAdd(p + 1, v.y);
    atomicAdd(p + 2, v.z);
    atomicAdd(p + 3, v.w);
}

// ---------------- WMMA fragment index helpers (16-bit, wave32) ----------------
// A (16x32, MxK): lane L holds row M=L%16; element i of v16bf maps to
//   K = 8*(L>=16) + i + (i>=8 ? 8 : 0)        (per ISA 7.12.2 table)
// B (32x16, KxN): lane L holds column N=L%16; element i maps to
//   K = 8*(L>=16) + i + (i>=8 ? 8 : 0)        (same as A)
// C/D (16x16 f32): VGPR r, lane L -> (M = r + 8*(L>=16), N = L%16)

__device__ __forceinline__ int a_k(int hs, int i) {
    return 8 * hs + i + ((i >= 8) ? 8 : 0);
}
__device__ __forceinline__ int b_k(int hs, int i) {
    return 8 * hs + i + ((i >= 8) ? 8 : 0);
}

// ---------------- Phase 2a: h = relu(mean@W1l^T + b1 + x@W1r^T) ----------------
// Block = one row group (8 row tiles); wave = one of 8 column tiles.
// B fragments for the wave's column live in registers for all 8 row tiles.

__global__ void __launch_bounds__(256)
sage_layer1(const float* __restrict__ acc, const float* __restrict__ counts,
            const float* __restrict__ x,
            const float* __restrict__ W1l, const float* __restrict__ b1,
            const float* __restrict__ W1r,
            float* __restrict__ h) {
    const int lane    = threadIdx.x & 31;
    const int colTile = threadIdx.x >> 5;       // 8 waves -> 8 column tiles
    const int l16 = lane & 15;
    const int hs  = lane >> 4;
    const int bcol = colTile * 16 + l16;

    // Weight-stationary: build B fragments once.
    v16bf bL[4], bR[4];
#pragma unroll
    for (int kb = 0; kb < 4; ++kb) {
#pragma unroll
        for (int i = 0; i < 16; ++i) {
            const int kB = kb * 32 + b_k(hs, i);
            bL[kb][i] = (__bf16)(W1l[bcol * DIM_IN + kB]);   // B[k][j] = W[j][k]
            bR[kb][i] = (__bf16)(W1r[bcol * DIM_IN + kB]);
        }
    }
    const float bias = b1[bcol];

    for (int rt = 0; rt < ROWS_PER_WAVE; ++rt) {
        const int rowTile = blockIdx.x * ROWS_PER_WAVE + rt;
        if (rowTile >= ROW_TILES) break;        // wave-uniform
        const int arow = rowTile * 16 + l16;

        // Prefetch next row tile's A rows (global_prefetch_b8).
        if (rowTile + 1 < ROW_TILES) {
            __builtin_prefetch(&acc[(arow + 16) * DIM_IN], 0, 1);
            __builtin_prefetch(&x[(arow + 16) * DIM_IN], 0, 1);
        }

        const float invc = 1.0f / fmaxf(counts[arow], 1.0f);
        v8f c = {};
#pragma unroll
        for (int kb = 0; kb < 4; ++kb) {        // K = 128 in 4 slabs of 32
            v16bf aM, aX;
#pragma unroll
            for (int i = 0; i < 16; ++i) {
                const int kA = kb * 32 + a_k(hs, i);
                aM[i] = (__bf16)(acc[arow * DIM_IN + kA] * invc);   // fused mean
                aX[i] = (__bf16)(x[arow * DIM_IN + kA]);
            }
            c = __builtin_amdgcn_wmma_f32_16x16x32_bf16(false, aM, false, bL[kb],
                                                        (short)0, c, false, false);
            c = __builtin_amdgcn_wmma_f32_16x16x32_bf16(false, aX, false, bR[kb],
                                                        (short)0, c, false, false);
        }
#pragma unroll
        for (int r = 0; r < 8; ++r) {
            const int grow = rowTile * 16 + 8 * hs + r;
            float v = c[r] + bias;
            h[grow * DIM_HID + bcol] = (v > 0.0f) ? v : 0.0f;       // fused ReLU
        }
    }
}

// ---------------- Phase 2b: out = mean2@W2l^T + b2 + h@W2r^T ----------------
// Global wave id -> (colTile in 0..2, rowGroup); each wave streams 8 row tiles.

#define L2_WAVE_TILES (3 * ((ROW_TILES + ROWS_PER_WAVE - 1) / ROWS_PER_WAVE))

__global__ void __launch_bounds__(256)
sage_layer2(const float* __restrict__ acc, const float* __restrict__ counts,
            const float* __restrict__ hin,
            const float* __restrict__ W2l, const float* __restrict__ b2,
            const float* __restrict__ W2r,
            float* __restrict__ out) {
    const int lane = threadIdx.x & 31;
    const int w    = blockIdx.x * 8 + (threadIdx.x >> 5);
    if (w >= L2_WAVE_TILES) return;             // wave-uniform
    const int colTile  = w % 3;                 // 40 cols -> 3 tiles, last partial
    const int rowGroup = w / 3;
    const int l16 = lane & 15;
    const int hs  = lane >> 4;
    const int bcol = colTile * 16 + l16;
    const bool colOK = (bcol < DIM_OUT);
    const int jc = colOK ? bcol : 0;            // clamp, not branch: EXEC stays full
    const float cmask = colOK ? 1.0f : 0.0f;

    v16bf bL[4], bR[4];
#pragma unroll
    for (int kb = 0; kb < 4; ++kb) {
#pragma unroll
        for (int i = 0; i < 16; ++i) {
            const int kB = kb * 32 + b_k(hs, i);
            bL[kb][i] = (__bf16)(W2l[jc * DIM_HID + kB] * cmask);
            bR[kb][i] = (__bf16)(W2r[jc * DIM_HID + kB] * cmask);
        }
    }
    const float bias = colOK ? b2[bcol] : 0.0f;

    for (int rt = 0; rt < ROWS_PER_WAVE; ++rt) {
        const int rowTile = rowGroup * ROWS_PER_WAVE + rt;
        if (rowTile >= ROW_TILES) break;        // wave-uniform
        const int arow = rowTile * 16 + l16;

        if (rowTile + 1 < ROW_TILES) {
            __builtin_prefetch(&acc[(arow + 16) * DIM_HID], 0, 1);
            __builtin_prefetch(&hin[(arow + 16) * DIM_HID], 0, 1);
        }

        const float invc = 1.0f / fmaxf(counts[arow], 1.0f);
        v8f c = {};
#pragma unroll
        for (int kb = 0; kb < 4; ++kb) {
            v16bf aM, aH;
#pragma unroll
            for (int i = 0; i < 16; ++i) {
                const int kA = kb * 32 + a_k(hs, i);
                aM[i] = (__bf16)(acc[arow * DIM_HID + kA] * invc);
                aH[i] = (__bf16)(hin[arow * DIM_HID + kA]);
            }
            c = __builtin_amdgcn_wmma_f32_16x16x32_bf16(false, aM, false, bL[kb],
                                                        (short)0, c, false, false);
            c = __builtin_amdgcn_wmma_f32_16x16x32_bf16(false, aH, false, bR[kb],
                                                        (short)0, c, false, false);
        }

        if (colOK) {
#pragma unroll
            for (int r = 0; r < 8; ++r) {
                const int grow = rowTile * 16 + 8 * hs + r;
                out[grow * DIM_OUT + bcol] = c[r] + bias;
            }
        }
    }
}

// ---------------- Launch ----------------

extern "C" void kernel_launch(void* const* d_in, const int* in_sizes, int n_in,
                              void* d_out, int out_size, void* d_ws, size_t ws_size,
                              hipStream_t stream) {
    const float*     x   = (const float*)d_in[0];
    const long long* ei  = (const long long*)d_in[1];   // int64 per reference
    const long long* src = ei;                          // edge_index[0]
    const long long* dst = ei + N_EDGES;                // edge_index[1]
    const float* W1l = (const float*)d_in[2];
    const float* b1  = (const float*)d_in[3];
    const float* W1r = (const float*)d_in[4];
    const float* W2l = (const float*)d_in[5];
    const float* b2  = (const float*)d_in[6];
    const float* W2r = (const float*)d_in[7];
    float* out = (float*)d_out;

    char* ws = (char*)d_ws;
    float* counts = (float*)ws;                                   // N floats
    size_t off1 = ((size_t)N_NODES * sizeof(float) + 255) & ~(size_t)255;
    float* acc = (float*)(ws + off1);                             // N*128 floats
    size_t off2 = off1 + (size_t)N_NODES * DIM_IN * sizeof(float);
    float* h = (float*)(ws + off2);                               // N*128 floats

    const int scatterThreads = N_EDGES * 32;
    const int rowGroups = (ROW_TILES + ROWS_PER_WAVE - 1) / ROWS_PER_WAVE;   // 391

    // Layer 1
    hipMemsetAsync(counts, 0, (size_t)N_NODES * sizeof(float), stream);
    hipMemsetAsync(acc, 0, (size_t)N_NODES * DIM_IN * sizeof(float), stream);
    count_kernel<<<(N_EDGES + 255) / 256, 256, 0, stream>>>(dst, counts);
    scatter_kernel<<<(scatterThreads + 255) / 256, 256, 0, stream>>>(src, dst, x, acc);
    sage_layer1<<<rowGroups, 256, 0, stream>>>(acc, counts, x, W1l, b1, W1r, h);

    // Layer 2 (reuse accumulator)
    hipMemsetAsync(acc, 0, (size_t)N_NODES * DIM_HID * sizeof(float), stream);
    scatter_kernel<<<(scatterThreads + 255) / 256, 256, 0, stream>>>(src, dst, h, acc);
    sage_layer2<<<(L2_WAVE_TILES + 7) / 8, 256, 0, stream>>>(acc, counts, h,
                                                             W2l, b2, W2r, out);
}
